// DDSCTransformer_9414568313157
// MI455X (gfx1250) — matrix operand order repaired and checked by reading, not compiled, not run on hardware
//
#include <hip/hip_runtime.h>
#include <hip/hip_bf16.h>
#include <math.h>

#define B_   8
#define L_   2048
#define C_   32
#define CH_  256
#define H_   8
#define D_   6
#define CHUNK 256

typedef __attribute__((ext_vector_type(16))) __bf16          v16bf;
typedef __attribute__((ext_vector_type(16))) unsigned short  v16u;
typedef __attribute__((ext_vector_type(8)))  float           v8f;
typedef __attribute__((address_space(3))) char               lds_char;

__device__ __forceinline__ unsigned short f2bf(float f) {
  unsigned int u = __float_as_uint(f);
  u += 0x7FFFu + ((u >> 16) & 1u);   // round-to-nearest-even
  return (unsigned short)(u >> 16);
}

// CDNA5 async memory->LDS copy (16B per lane), tracked by ASYNCcnt.
__device__ __forceinline__ void async_ld16(lds_char* lds, const void* g) {
  asm volatile("global_load_async_to_lds_b128 %0, %1, off"
               :: "v"(lds), "v"(g) : "memory");
}
__device__ __forceinline__ void wait_async0() {
  asm volatile("s_wait_asynccnt 0" ::: "memory");
}

// ---------------- encoder: h = enc_W(32x6) @ x(B,6,L) + b ----------------
__global__ void encoder_kernel(const float* __restrict__ x,
                               const float* __restrict__ W,
                               const float* __restrict__ bias,
                               float* __restrict__ out) {
  int idx = blockIdx.x * blockDim.x + threadIdx.x;     // B*32*L
  int l = idx & (L_ - 1);
  int c = (idx >> 11) & (C_ - 1);
  int b = idx >> 16;
  float acc = bias[c];
#pragma unroll
  for (int i = 0; i < 6; ++i)
    acc += W[c * 6 + i] * x[(size_t)(b * 6 + i) * L_ + l];
  out[idx] = acc;
}

// ---------------- pointwise: t(B,256,L) = pw(256x32) @ x(B,32,L) ----------------
__global__ __launch_bounds__(128) void pointwise_kernel(const float* __restrict__ x,
                                                        const float* __restrict__ pw,
                                                        float* __restrict__ t) {
  __shared__ float xs[C_ * 128];
  __shared__ float wls[CH_ * C_];
  int tid = threadIdx.x;
  int lt = blockIdx.x & 15;            // L/128 = 16 tiles
  int b  = blockIdx.x >> 4;
  int l0 = lt * 128;
  for (int i = tid; i < C_ * 128; i += 128) {
    int c = i >> 7, l = i & 127;
    xs[c * 128 + l] = x[(size_t)(b * C_ + c) * L_ + l0 + l];
  }
  for (int i = tid; i < CH_ * C_; i += 128) wls[i] = pw[i];
  __syncthreads();
  for (int ch = 0; ch < CH_; ++ch) {
    float acc = 0.f;
#pragma unroll
    for (int c = 0; c < C_; ++c) acc += wls[ch * C_ + c] * xs[c * 128 + tid];
    t[(size_t)(b * CH_ + ch) * L_ + l0 + tid] = acc;
  }
}

// ------ depthwise 3 & 15 with softmax gate; emits bf16 for WMMA consumption ------
__global__ void dwgate_kernel(const float* __restrict__ t,
                              const float* __restrict__ w3,
                              const float* __restrict__ w15,
                              const float* __restrict__ gate,
                              unsigned short* __restrict__ out) {
  int idx = blockIdx.x * blockDim.x + threadIdx.x;     // B*256*L
  int l  = idx & (L_ - 1);
  int ch = (idx >> 11) & (CH_ - 1);
  int b  = idx >> 19;
  const float* row = t + (size_t)(b * CH_ + ch) * L_;
  float g0 = __expf(gate[0]), g1 = __expf(gate[1]);
  float gs = 1.0f / (g0 + g1); g0 *= gs; g1 *= gs;
  float a3 = 0.f;
#pragma unroll
  for (int k = 0; k < 3; ++k) {
    int p = l + k - 1;
    if (p >= 0 && p < L_) a3 += row[p] * w3[ch * 3 + k];
  }
  float a15 = 0.f;
#pragma unroll
  for (int k = 0; k < 15; ++k) {
    int p = l + k - 7;
    if (p >= 0 && p < L_) a15 += row[p] * w15[ch * 15 + k];
  }
  out[idx] = f2bf(g0 * a3 + g1 * a15);
}

// ---------------- flash attention per (b,h), bf16 WMMA, fp32 accum ----------------
__global__ __launch_bounds__(128) void flash_attn_kernel(
    const unsigned short* __restrict__ Qb,
    const unsigned short* __restrict__ Kb,
    const unsigned short* __restrict__ Vb,
    float* __restrict__ Ob) {
  __shared__ unsigned short sK[C_ * CHUNK];
  __shared__ unsigned short sV[C_ * CHUNK];
  const int tid  = threadIdx.x;
  const int lane = tid & 31;
  const int wave = tid >> 5;
  const int n  = lane & 15;            // column / A-row index
  const int hi = lane >> 4;            // half-wave select
  const int bh = blockIdx.y;
  const int b  = bh >> 3;
  const int h0 = (bh & 7) * C_;
  const int qt = (blockIdx.x * 4 + wave) * 16;
  const float scale2 = 0.17677669529663688f;  // C^-0.5 (scale applied to K and Q)

  // Q tile, B-matrix layout: contraction c = hi*16 + j
  v16u qu;
#pragma unroll
  for (int j = 0; j < 16; ++j) {
    int c = hi * 16 + j;
    qu[j] = Qb[(size_t)(b * CH_ + h0 + c) * L_ + qt + n];
  }
  v16bf qv = __builtin_bit_cast(v16bf, qu);

  float m_run = -3.0e38f, l_run = 0.0f;
  v8f O0 = {}; v8f O1 = {}; v8f zc = {};

  const char* Kbase = (const char*)Kb;
  const char* Vbase = (const char*)Vb;
  lds_char* sKl = (lds_char*)sK;
  lds_char* sVl = (lds_char*)sV;

  for (int kc = 0; kc < L_; kc += CHUNK) {
    __syncthreads();
    // Async stage K/V chunk [32 x CHUNK] bf16 into LDS: 1024 16B segments each.
    size_t rowbase = (size_t)(b * CH_ + h0) * L_ + kc;    // element offset of c=0
#pragma unroll
    for (int i = 0; i < 8; ++i) {
      int s = tid + i * 128;            // segment id 0..1023
      int c = s >> 5, seg = s & 31;     // 32 segments (16B) per 512B row
      size_t gofs = (rowbase + (size_t)c * L_) * 2 + (size_t)seg * 16;
      unsigned lofs = (unsigned)(c * (CHUNK * 2) + seg * 16);
      async_ld16(sKl + lofs, Kbase + gofs);
      async_ld16(sVl + lofs, Vbase + gofs);
    }
    wait_async0();
    __syncthreads();

#pragma unroll 1
    for (int kk = 0; kk < CHUNK; kk += 32) {
      // A = K^T tiles: row m = n -> key kk+n / kk+16+n; element j -> channel c
      v16u a0u, a1u;
#pragma unroll
      for (int j = 0; j < 16; ++j) {
        int c = hi * 16 + j;
        a0u[j] = sK[c * CHUNK + kk + n];
        a1u[j] = sK[c * CHUNK + kk + 16 + n];
      }
      v8f S0 = __builtin_amdgcn_wmma_f32_16x16x32_bf16(
          false, __builtin_bit_cast(v16bf, a0u), false, qv, (short)0, zc, false, false);
      v8f S1 = __builtin_amdgcn_wmma_f32_16x16x32_bf16(
          false, __builtin_bit_cast(v16bf, a1u), false, qv, (short)0, zc, false, false);

      // ---- online softmax over the 32 keys of this step (per column n) ----
      float s0[8], s1[8];
      float mloc = -3.0e38f;
#pragma unroll
      for (int r = 0; r < 8; ++r) {
        s0[r] = S0[r] * scale2; s1[r] = S1[r] * scale2;
        mloc = fmaxf(mloc, fmaxf(s0[r], s1[r]));
      }
      float mchunk = fmaxf(mloc, __shfl_xor(mloc, 16, 32));
      float mnew   = fmaxf(m_run, mchunk);
      float alpha  = __expf(m_run - mnew);
      float p0[8], p1[8], psum = 0.f;
#pragma unroll
      for (int r = 0; r < 8; ++r) {
        p0[r] = __expf(s0[r] - mnew);
        p1[r] = __expf(s1[r] - mnew);
        psum += p0[r] + p1[r];
      }
      l_run = l_run * alpha + psum + __shfl_xor(psum, 16, 32);
      m_run = mnew;

      // ---- repack P (C-layout -> B-layout, k = hi*16 + j) ----
      // Each lane needs only the partner's p0 (hi=0) or p1 (hi=1): one 8-value swap.
      float recv[8];
#pragma unroll
      for (int r = 0; r < 8; ++r) {
        float sendv = hi ? p0[r] : p1[r];
        recv[r] = __shfl_xor(sendv, 16, 32);
      }
      v16u pu;
#pragma unroll
      for (int j = 0; j < 16; ++j) {
        float pj = (j < 8) ? (hi ? recv[j] : p0[j])
                           : (hi ? p1[j - 8] : recv[j - 8]);
        pu[j] = f2bf(pj);
      }
      v16bf pv = __builtin_bit_cast(v16bf, pu);

      // ---- O halves: A = V rows c=n / 16+n, contraction = 32 keys of step ----
      v16u v0u, v1u;
#pragma unroll
      for (int j = 0; j < 16; ++j) {
        int kj = hi * 16 + j;
        v0u[j] = sV[n * CHUNK + kk + kj];
        v1u[j] = sV[(16 + n) * CHUNK + kk + kj];
      }
#pragma unroll
      for (int r = 0; r < 8; ++r) { O0[r] *= alpha; O1[r] *= alpha; }
      O0 = __builtin_amdgcn_wmma_f32_16x16x32_bf16(
          false, __builtin_bit_cast(v16bf, v0u), false, pv, (short)0, O0, false, false);
      O1 = __builtin_amdgcn_wmma_f32_16x16x32_bf16(
          false, __builtin_bit_cast(v16bf, v1u), false, pv, (short)0, O1, false, false);
    }
  }

  float inv = 1.0f / l_run;
#pragma unroll
  for (int r = 0; r < 8; ++r) {
    int c = hi * 8 + r;  // C/D layout: lanes 0-15 rows 0-7, lanes 16-31 rows 8-15
    Ob[(size_t)(b * CH_ + h0 + c) * L_ + qt + n]      = O0[r] * inv;
    Ob[(size_t)(b * CH_ + h0 + 16 + c) * L_ + qt + n] = O1[r] * inv;
  }
}

// -------- uni: y = uni_W(32x256) @ o + uni_b + residual --------
__global__ void uni_res_kernel(const float* __restrict__ o,
                               const float* __restrict__ W,
                               const float* __restrict__ bias,
                               const float* __restrict__ resid,
                               float* __restrict__ y) {
  int idx = blockIdx.x * blockDim.x + threadIdx.x;     // B*32*L
  int l = idx & (L_ - 1);
  int c = (idx >> 11) & (C_ - 1);
  int b = idx >> 16;
  float acc = bias[c];
  const float* wr = W + c * CH_;
  const float* ob = o + (size_t)b * CH_ * L_ + l;
#pragma unroll 4
  for (int oc = 0; oc < CH_; ++oc) acc += wr[oc] * ob[(size_t)oc * L_];
  y[idx] = acc + resid[idx];
}

// -------- InstanceNorm1d over L (biased var), in-place --------
__global__ void inorm_kernel(float* __restrict__ xio,
                             const float* __restrict__ g,
                             const float* __restrict__ bta) {
  __shared__ float red[256];
  int b = blockIdx.x >> 5;
  int c = blockIdx.x & 31;
  float* row = xio + (size_t)(b * C_ + c) * L_;
  int tid = threadIdx.x;
  float lv[8], s = 0.f;
#pragma unroll
  for (int i = 0; i < 8; ++i) { lv[i] = row[tid + i * 256]; s += lv[i]; }
  red[tid] = s; __syncthreads();
  for (int off = 128; off > 0; off >>= 1) {
    if (tid < off) red[tid] += red[tid + off];
    __syncthreads();
  }
  float mean = red[0] * (1.0f / L_);
  __syncthreads();
  float vs = 0.f;
#pragma unroll
  for (int i = 0; i < 8; ++i) { float d = lv[i] - mean; vs += d * d; }
  red[tid] = vs; __syncthreads();
  for (int off = 128; off > 0; off >>= 1) {
    if (tid < off) red[tid] += red[tid + off];
    __syncthreads();
  }
  float inv = rsqrtf(red[0] * (1.0f / L_) + 1e-5f) * g[c];
  float bb = bta[c];
#pragma unroll
  for (int i = 0; i < 8; ++i) row[tid + i * 256] = (lv[i] - mean) * inv + bb;
}

// -------- FFN --------
__global__ void ffn1_kernel(const float* __restrict__ xin,
                            const float* __restrict__ W1,
                            const float* __restrict__ b1,
                            float* __restrict__ hbuf) {
  int idx = blockIdx.x * blockDim.x + threadIdx.x;     // B*128*L
  int l = idx & (L_ - 1);
  int f = (idx >> 11) & 127;
  int b = idx >> 18;
  float acc = b1[f];
#pragma unroll
  for (int c = 0; c < C_; ++c)
    acc += W1[f * C_ + c] * xin[(size_t)(b * C_ + c) * L_ + l];
  hbuf[idx] = fmaxf(acc, 0.f);
}

__global__ void ffn2_kernel(const float* __restrict__ hbuf,
                            const float* __restrict__ W2,
                            const float* __restrict__ b2,
                            const float* __restrict__ resid,
                            float* __restrict__ out) {
  int idx = blockIdx.x * blockDim.x + threadIdx.x;     // B*32*L
  int l = idx & (L_ - 1);
  int c = (idx >> 11) & (C_ - 1);
  int b = idx >> 16;
  float acc = b2[c];
#pragma unroll 4
  for (int f = 0; f < 128; ++f)
    acc += W2[c * 128 + f] * hbuf[(size_t)(b * 128 + f) * L_ + l];
  out[idx] = acc + resid[idx];
}

// -------- classifier + sigmoid --------
__global__ void cls_kernel(const float* __restrict__ xin,
                           const float* __restrict__ W,
                           const float* __restrict__ bias,
                           float* __restrict__ out) {
  int idx = blockIdx.x * blockDim.x + threadIdx.x;     // B*L
  int l = idx & (L_ - 1);
  int b = idx >> 11;
  float acc = bias[0];
#pragma unroll
  for (int c = 0; c < C_; ++c)
    acc += W[c] * xin[(size_t)(b * C_ + c) * L_ + l];
  out[idx] = 1.0f / (1.0f + __expf(-acc));
}

extern "C" void kernel_launch(void* const* d_in, const int* in_sizes, int n_in,
                              void* d_out, int out_size, void* d_ws, size_t ws_size,
                              hipStream_t stream) {
  (void)in_sizes; (void)n_in; (void)out_size; (void)ws_size;
  const float* x      = (const float*)d_in[0];
  const float* enc_W  = (const float*)d_in[1];
  const float* enc_b  = (const float*)d_in[2];
  const float* pw[3]   = { (const float*)d_in[3],  (const float*)d_in[7],  (const float*)d_in[11] };
  const float* dw3[3]  = { (const float*)d_in[4],  (const float*)d_in[8],  (const float*)d_in[12] };
  const float* dw15[3] = { (const float*)d_in[5],  (const float*)d_in[9],  (const float*)d_in[13] };
  const float* gate[3] = { (const float*)d_in[6],  (const float*)d_in[10], (const float*)d_in[14] };
  const float* uni_W  = (const float*)d_in[15];
  const float* uni_b  = (const float*)d_in[16];
  const float* n1_g   = (const float*)d_in[17];
  const float* n1_b   = (const float*)d_in[18];
  const float* n2_g   = (const float*)d_in[19];
  const float* n2_b   = (const float*)d_in[20];
  const float* ffn_W1 = (const float*)d_in[21];
  const float* ffn_b1 = (const float*)d_in[22];
  const float* ffn_W2 = (const float*)d_in[23];
  const float* ffn_b2 = (const float*)d_in[24];
  const float* cls_W  = (const float*)d_in[25];
  const float* cls_b  = (const float*)d_in[26];

  char* ws = (char*)d_ws;
  float* A  = (float*)(ws);                               // [B,32,L]  2 MB  (residual stream)
  float* Bv = (float*)(ws + ((size_t)2  << 20));          // [B,32,L]  2 MB
  float* Hb = (float*)(ws + ((size_t)4  << 20));          // [B,128,L] 8 MB
  float* T  = (float*)(ws + ((size_t)12 << 20));          // [B,256,L] 16 MB (pw tmp / attn out)
  unsigned short* Qb = (unsigned short*)(ws + ((size_t)28 << 20)); // bf16 8 MB
  unsigned short* Kb = (unsigned short*)(ws + ((size_t)36 << 20));
  unsigned short* Vb = (unsigned short*)(ws + ((size_t)44 << 20));
  unsigned short* QKV[3] = { Qb, Kb, Vb };

  encoder_kernel<<<dim3(B_ * C_ * L_ / 256), 256, 0, stream>>>(x, enc_W, enc_b, A);

  for (int d = 0; d < D_; ++d) {
    for (int t3 = 0; t3 < 3; ++t3) {
      pointwise_kernel<<<dim3(B_ * 16), 128, 0, stream>>>(A, pw[t3] + (size_t)d * CH_ * C_, T);
      dwgate_kernel<<<dim3(B_ * CH_ * L_ / 256), 256, 0, stream>>>(
          T, dw3[t3] + (size_t)d * CH_ * 3, dw15[t3] + (size_t)d * CH_ * 15,
          gate[t3] + (size_t)d * 2, QKV[t3]);
    }
    flash_attn_kernel<<<dim3(L_ / 64, B_ * H_), 128, 0, stream>>>(Qb, Kb, Vb, T);
    uni_res_kernel<<<dim3(B_ * C_ * L_ / 256), 256, 0, stream>>>(
        T, uni_W + (size_t)d * C_ * CH_, uni_b + (size_t)d * C_, A, Bv);
    inorm_kernel<<<dim3(B_ * C_), 256, 0, stream>>>(Bv, n1_g + d * C_, n1_b + d * C_);
    ffn1_kernel<<<dim3(B_ * 128 * L_ / 256), 256, 0, stream>>>(
        Bv, ffn_W1 + (size_t)d * 128 * C_, ffn_b1 + (size_t)d * 128, Hb);
    ffn2_kernel<<<dim3(B_ * C_ * L_ / 256), 256, 0, stream>>>(
        Hb, ffn_W2 + (size_t)d * C_ * 128, ffn_b2 + (size_t)d * C_, Bv, A);
    inorm_kernel<<<dim3(B_ * C_), 256, 0, stream>>>(A, n2_g + d * C_, n2_b + d * C_);
  }

  cls_kernel<<<dim3(B_ * L_ / 256), 256, 0, stream>>>(A, cls_W, cls_b, (float*)d_out);
}
